// MNRDEFunc_84808424227321
// MI455X (gfx1250) — matrix operand order repaired and checked by reading, not compiled, not run on hardware
//
#include <hip/hip_runtime.h>
#include <hip/hip_bf16.h>

// ---------------------------------------------------------------------------
// MNRDE log-ODE vector field: fused forward MLP + 8-direction JVP + Lyndon
// bracket contraction, built around v_wmma_f32_16x16x32_f16 (fp32 accum).
// One workgroup (8 wave32s, 256 threads) owns 8 batch elements end-to-end in
// LDS (~258KB of the 320KB/WGP). Weights converted once to f16 row-major
// (== WMMA A-operand friendly) into the workspace.
// Epilogues use fast TRANS32 ops (v_exp/v_log/v_rcp) branch-free so EXEC stays
// uniform and VALU work co-executes with the matrix pipeline.
// ---------------------------------------------------------------------------

typedef _Float16 f16;
typedef __attribute__((ext_vector_type(16))) _Float16 v16h;
typedef __attribute__((ext_vector_type(8)))  _Float16 v8h;
typedef __attribute__((ext_vector_type(8)))  float    v8f;

#define NW 8          // waves per block (256 threads, wave32)
#define D8 8          // input path dim
#define NSTATE 256    // cde state dim
#define HDIM 512      // hidden dim
#define ODIM 2048     // D*N output of MLP
#define EPB 8         // batch elements per workgroup
#define JCOLS 64      // EPB * D8 tangent columns

union V16U { uint4 q[2]; v16h v; };
union V8U  { uint4 q;    v8h  h; };

// Lyndon pair index for (i<j), D=8, triu order
__device__ __forceinline__ int pidx(int i, int j) {
  return i * 7 - (i * (i - 1)) / 2 + (j - i - 1);
}

__device__ __forceinline__ float fast_rcp(float x) {
  return __builtin_amdgcn_rcpf(x);
}

// A operand (16x32 f16): weights row-major in global. Lane<16 holds row m=lane,
// K {k0..k0+7, k0+16..k0+23}; lane>=16 holds row m=lane-16, K {+8, +24}.
__device__ __forceinline__ v16h load_a_global(const f16* __restrict__ W, int Kdim,
                                              int row, int k0, int laneHi) {
  const f16* p = W + (size_t)row * Kdim + k0 + (laneHi ? 8 : 0);
  V16U u;
  u.q[0] = *(const uint4*)p;          // 8 halfs
  u.q[1] = *(const uint4*)(p + 16);   // 8 halfs, +16 K
  return u.v;
}

// B operand (32x16 f16): column-major (K contiguous) in LDS. Lane<16 = col,
// K {k0..k0+7, k0+16..k0+23}; lane>=16 = col, K {+8, +24}.
__device__ __forceinline__ v16h load_b_lds(const f16* B, int Kdim,
                                           int col, int k0, int laneHi) {
  const f16* p = B + col * Kdim + k0 + (laneHi ? 8 : 0);
  V16U u;
  u.q[0] = *(const uint4*)p;
  u.q[1] = *(const uint4*)(p + 16);
  return u.v;
}

__device__ __forceinline__ v8f wmma_step(v16h a, v16h b, v8f acc) {
  return __builtin_amdgcn_wmma_f32_16x16x32_f16(false, a, false, b,
                                                (short)0, acc, false, false);
}

// Forward hidden layer: X_out = softplus(W @ X_in + b), S_out = sigmoid(z).
__device__ __forceinline__ void fwd_layer(const f16* __restrict__ W,
                                          const float* __restrict__ bias,
                                          const f16* Bin, int Kin,
                                          f16* Xout, f16* Sout,
                                          int wave, int lmod, int laneHi) {
  for (int t = wave; t < (HDIM >> 4); t += NW) {   // 32 M-tiles
    int mt = t << 4;
    v8f acc = {};
    for (int k0 = 0; k0 < Kin; k0 += 32) {
      v16h a = load_a_global(W, Kin, mt + lmod, k0, laneHi);
      v16h b = load_b_lds(Bin, Kin, lmod, k0, laneHi);
      acc = wmma_step(a, b, acc);
    }
    int e  = lmod;                       // batch column
    int r0 = mt + (laneHi << 3);         // first output row of this lane
    if (e < EPB) {
      float4 blo = *(const float4*)(bias + r0);
      float4 bhi = *(const float4*)(bias + r0 + 4);
      float bz[8] = {blo.x, blo.y, blo.z, blo.w, bhi.x, bhi.y, bhi.z, bhi.w};
      V8U xo, so;
#pragma unroll
      for (int v = 0; v < 8; ++v) {
        float z  = acc[v] + bz[v];
        float ex = __expf(z);
        // softplus: exact-to-fp32 for z>20; log(1+e^z) saturates fine below
        float sp = (z > 20.f) ? z : __logf(1.f + ex);
        // sigmoid = 1 - 1/(1+e^z): ex=inf -> 1, ex=0 -> 0 (branch-free)
        float sg = 1.f - fast_rcp(1.f + ex);
        xo.h[v] = (f16)sp;
        so.h[v] = (f16)sg;
      }
      *(uint4*)(Xout + e * HDIM + r0) = xo.q;
      *(uint4*)(Sout + e * HDIM + r0) = so.q;
    }
  }
}

// Forward output layer: V = tanh(Wo @ X3 + bo) -> Vb (as tangent B operand),
// To = 1 - V^2 (tanh').
__device__ __forceinline__ void fwd_final(const f16* __restrict__ W,
                                          const float* __restrict__ bias,
                                          const f16* Bin, f16* Vb, f16* To,
                                          int wave, int lmod, int laneHi) {
  for (int t = wave; t < (ODIM >> 4); t += NW) {   // 128 M-tiles
    int mt = t << 4;
    v8f acc = {};
    for (int k0 = 0; k0 < HDIM; k0 += 32) {
      v16h a = load_a_global(W, HDIM, mt + lmod, k0, laneHi);
      v16h b = load_b_lds(Bin, HDIM, lmod, k0, laneHi);
      acc = wmma_step(a, b, acc);
    }
    int e  = lmod;
    int r0 = mt + (laneHi << 3);
    if (e < EPB) {
      float4 blo = *(const float4*)(bias + r0);
      float4 bhi = *(const float4*)(bias + r0 + 4);
      float bz[8] = {blo.x, blo.y, blo.z, blo.w, bhi.x, bhi.y, bhi.z, bhi.w};
      V8U vv, tt;
#pragma unroll
      for (int v = 0; v < 8; ++v) {
        float z  = acc[v] + bz[v];
        // tanh(z) = 1 - 2/(e^{2z}+1): branch-free, saturates to +/-1
        float e2 = __expf(2.f * z);
        float th = 1.f - 2.f * fast_rcp(e2 + 1.f);
        vv.h[v] = (f16)th;
        tt.h[v] = (f16)(1.f - th * th);
      }
      int i = r0 >> 8, n0 = r0 & 255;     // row r = i*256 + n
      *(uint4*)(Vb + (e * D8 + i) * NSTATE + n0) = vv.q;   // tangent column layout
      *(uint4*)(To + e * ODIM + r0) = tt.q;
    }
  }
}

// JVP hidden layer: U_out = S_l(e) .* (W @ U_in), 64 columns (8 elems x 8 dirs).
__device__ __forceinline__ void jvp_layer(const f16* __restrict__ W,
                                          const f16* Uin, int Kin,
                                          const f16* S, f16* Uout,
                                          int wave, int lmod, int laneHi) {
  for (int t = wave; t < (HDIM >> 4) * 4; t += NW) {   // 32 M-tiles x 4 N-tiles
    int mt = (t >> 2) << 4;
    int nt = (t & 3) << 4;
    v8f acc = {};
    for (int k0 = 0; k0 < Kin; k0 += 32) {
      v16h a = load_a_global(W, Kin, mt + lmod, k0, laneHi);
      v16h b = load_b_lds(Uin, Kin, nt + lmod, k0, laneHi);
      acc = wmma_step(a, b, acc);
    }
    int c  = nt + lmod;
    int e  = c >> 3;
    int r0 = mt + (laneHi << 3);
    V8U sv; sv.q = *(const uint4*)(S + e * HDIM + r0);
    V8U uo;
#pragma unroll
    for (int v = 0; v < 8; ++v)
      uo.h[v] = (f16)(acc[v] * (float)sv.h[v]);
    *(uint4*)(Uout + c * HDIM + r0) = uo.q;
  }
}

// JVP output layer fused with bracket contraction:
// J[e,k,i,n] = To[e,i*256+n] * (Wo @ U3)[i*256+n, col(e,k)]
// outacc[e,n] += coeff[e,k,i] * J
__device__ __forceinline__ void jvp_final(const f16* __restrict__ W,
                                          const f16* Uin, const f16* To,
                                          const float* coeff, float* outacc,
                                          int wave, int lmod, int laneHi) {
  for (int t = wave; t < (ODIM >> 4) * 4; t += NW) {   // 128 x 4 tiles
    int mt = (t >> 2) << 4;
    int nt = (t & 3) << 4;
    v8f acc = {};
    for (int k0 = 0; k0 < HDIM; k0 += 32) {
      v16h a = load_a_global(W, HDIM, mt + lmod, k0, laneHi);
      v16h b = load_b_lds(Uin, HDIM, nt + lmod, k0, laneHi);
      acc = wmma_step(a, b, acc);
    }
    int c  = nt + lmod;
    int e  = c >> 3;
    int k  = c & 7;
    int r0 = mt + (laneHi << 3);
    int i  = r0 >> 8, n0 = r0 & 255;
    float cf = coeff[e * 64 + k * 8 + i];
    if (cf != 0.f) {
      V8U tv; tv.q = *(const uint4*)(To + e * ODIM + r0);
#pragma unroll
      for (int v = 0; v < 8; ++v)
        atomicAdd(&outacc[e * NSTATE + n0 + v], cf * (float)tv.h[v] * acc[v]);
    }
  }
}

__global__ void __launch_bounds__(256, 1)
mnrde_fused_kernel(const float* __restrict__ h, const float* __restrict__ sig,
                   const f16* __restrict__ W1h, const float* __restrict__ b1,
                   const f16* __restrict__ W2h, const float* __restrict__ b2,
                   const f16* __restrict__ W3h, const float* __restrict__ b3,
                   const f16* __restrict__ Woh, const float* __restrict__ bo,
                   float* __restrict__ out) {
  extern __shared__ char smem[];
  f16* Xa = (f16*)smem;                  // 8192 halfs: X0 (16x256) then X2 (16x512)
  f16* Xb = Xa + 8192;                   // 8192 halfs: X1 / X3
  f16* S1 = Xb + 8192;                   // [8][512]
  f16* S2 = S1 + 4096;
  f16* S3 = S2 + 4096;
  f16* To = S3 + 4096;                   // [8][2048]
  f16* Vb = To + 16384;                  // [64 cols][256]  tangent B operand
  f16* Ua = Vb + 16384;                  // [64][512]
  f16* Ub = Ua + 32768;                  // [64][512]
  float* outacc = (float*)(Ub + 32768);  // [8][256]
  float* coeff  = outacc + 2048;         // [8][8 dir][8 field]

  const int tid    = threadIdx.x;
  const int lane   = tid & 31;
  const int wave   = tid >> 5;
  const int laneHi = lane >> 4;
  const int lmod   = lane & 15;
  const int ebase  = blockIdx.x * EPB;

  // ---- init: zero activation buffers (padding cols 8..15 stay zero forever)
  for (int idx = tid; idx < 8192; idx += 256) { Xa[idx] = (f16)0; Xb[idx] = (f16)0; }
  __syncthreads();
  // load h -> X0 (col-major [e][256]) and build +/- sig coefficient table
  for (int idx = tid; idx < EPB * NSTATE; idx += 256) {
    int e = idx >> 8, n = idx & 255;
    Xa[e * NSTATE + n] = (f16)h[(size_t)(ebase + e) * NSTATE + n];
  }
  for (int idx = tid; idx < EPB * 64; idx += 256) {
    int e = idx >> 6, k = (idx >> 3) & 7, i = idx & 7;
    const float* sg = sig + (size_t)(ebase + e) * 36;
    float c = 0.f;
    if (k < i)      c =  sg[8 + pidx(k, i)];
    else if (k > i) c = -sg[8 + pidx(i, k)];
    coeff[idx] = c;
  }
  __syncthreads();

  // ---- forward MLP ----
  fwd_layer(W1h, b1, Xa, NSTATE, Xb, S1, wave, lmod, laneHi);  __syncthreads();
  fwd_layer(W2h, b2, Xb, HDIM,   Xa, S2, wave, lmod, laneHi);  __syncthreads();
  fwd_layer(W3h, b3, Xa, HDIM,   Xb, S3, wave, lmod, laneHi);  __syncthreads();
  fwd_final(Woh, bo, Xb, Vb, To, wave, lmod, laneHi);          __syncthreads();

  // ---- level-1 term: outacc[e,n] = sum_i sig[e,i] * V[e,i,n]
  for (int idx = tid; idx < EPB * NSTATE; idx += 256) {
    int e = idx >> 8, n = idx & 255;
    const float* sg = sig + (size_t)(ebase + e) * 36;
    float s = 0.f;
#pragma unroll
    for (int i = 0; i < D8; ++i)
      s += sg[i] * (float)Vb[(e * D8 + i) * NSTATE + n];
    outacc[idx] = s;
  }
  __syncthreads();

  // ---- JVP in 8 directions (columns = V rows), chained GEMMs ----
  jvp_layer(W1h, Vb, NSTATE, S1, Ua, wave, lmod, laneHi);  __syncthreads();
  jvp_layer(W2h, Ua, HDIM,   S2, Ub, wave, lmod, laneHi);  __syncthreads();
  jvp_layer(W3h, Ub, HDIM,   S3, Ua, wave, lmod, laneHi);  __syncthreads();
  jvp_final(Woh, Ua, To, coeff, outacc, wave, lmod, laneHi);
  __syncthreads();

  // ---- write result
  for (int idx = tid; idx < EPB * NSTATE; idx += 256)
    out[(size_t)(ebase + (idx >> 8)) * NSTATE + (idx & 255)] = outacc[idx];
}

// fp32 -> f16 weight conversion (row-major preserved == A-operand layout)
__global__ void cvt_f32_to_f16(const float* __restrict__ src, f16* __restrict__ dst, int n) {
  int i = blockIdx.x * blockDim.x + threadIdx.x;
  if (i < n) dst[i] = (f16)src[i];
}

extern "C" void kernel_launch(void* const* d_in, const int* in_sizes, int n_in,
                              void* d_out, int out_size, void* d_ws, size_t ws_size,
                              hipStream_t stream) {
  (void)in_sizes; (void)n_in; (void)out_size; (void)ws_size;
  const float* h   = (const float*)d_in[0];
  const float* sig = (const float*)d_in[1];
  const float* W1  = (const float*)d_in[2];
  const float* b1  = (const float*)d_in[3];
  const float* W2  = (const float*)d_in[4];
  const float* b2  = (const float*)d_in[5];
  const float* W3  = (const float*)d_in[6];
  const float* b3  = (const float*)d_in[7];
  const float* Wo  = (const float*)d_in[8];
  const float* bo  = (const float*)d_in[9];

  f16* W1h = (f16*)d_ws;                  // 512*256
  f16* W2h = W1h + 512 * 256;             // 512*512
  f16* W3h = W2h + 512 * 512;             // 512*512
  f16* Woh = W3h + 512 * 512;             // 2048*512

  cvt_f32_to_f16<<<(512 * 256  + 255) / 256, 256, 0, stream>>>(W1, W1h, 512 * 256);
  cvt_f32_to_f16<<<(512 * 512  + 255) / 256, 256, 0, stream>>>(W2, W2h, 512 * 512);
  cvt_f32_to_f16<<<(512 * 512  + 255) / 256, 256, 0, stream>>>(W3, W3h, 512 * 512);
  cvt_f32_to_f16<<<(2048 * 512 + 255) / 256, 256, 0, stream>>>(Wo, Woh, 2048 * 512);

  // LDS: (8192+8192+3*4096+16384+16384+2*32768) halfs + (2048+512) floats
  size_t lds_bytes = 126976 * sizeof(f16) + 2560 * sizeof(float);  // 264192 B
  mnrde_fused_kernel<<<4096 / EPB, 256, lds_bytes, stream>>>(
      h, sig, W1h, b1, W2h, b2, W3h, b3, Woh, bo, (float*)d_out);
}
